// GraphAttentionLayer_60825326846768
// MI455X (gfx1250) — matrix operand order repaired and checked by reading, not compiled, not run on hardware
//
#include <hip/hip_runtime.h>

typedef __attribute__((ext_vector_type(2)))  float        v2f;
typedef __attribute__((ext_vector_type(8)))  float        v8f;
typedef __attribute__((ext_vector_type(16))) __bf16       v16bf;
typedef __attribute__((ext_vector_type(4)))  unsigned int v4u;

#define NB   8
#define NN   2048
#define KIN  256
#define DOUT 128
#define ROWS (NB * NN)   // 16384 flat rows (b*2048 + i)

__device__ __forceinline__ unsigned short f2bf(float x) {
  unsigned u = __float_as_uint(x);
  u += 0x7FFFu + ((u >> 16) & 1u);          // round-to-nearest-even
  return (unsigned short)(u >> 16);
}

union ABf { v4u q[2]; v16bf v; };

// ---------------------------------------------------------------------------
// Kernel 1: Wh = h @ W  via V_WMMA_F32_16X16X4_F32 (exact f32; feeds logits).
// One wave owns a 16x16 tile: block = 16 M-rows, 8 waves cover N=0..127.
// A layout (32-bit 16x4): lane l holds M=l&15, K = (l<16 ? {k,k+1} : {k+2,k+3}).
// ---------------------------------------------------------------------------
__global__ __launch_bounds__(256) void k_gemm_wh(
    const float* __restrict__ h, const float* __restrict__ W,
    float* __restrict__ Wh, unsigned short* __restrict__ Whb)
{
  const int lane = threadIdx.x & 31, wave = threadIdx.x >> 5;
  const int half = lane >> 4, ml = lane & 15;
  const int row = blockIdx.x * 16 + ml;      // A-matrix row
  const int n   = wave * 16 + ml;            // B/C column
  const int kb  = half * 2;
  v8f c = {};
  #pragma unroll 4
  for (int k = 0; k < KIN; k += 4) {
    v2f a, b;
    a.x = h[(size_t)row * KIN + k + kb];
    a.y = h[(size_t)row * KIN + k + kb + 1];
    b.x = W[(size_t)(k + kb)     * DOUT + n];
    b.y = W[(size_t)(k + kb + 1) * DOUT + n];
    c = __builtin_amdgcn_wmma_f32_16x16x4_f32(false, a, false, b,
                                              (short)0, c, false, false);
  }
  // C layout: lanes 0-15 -> M = r, lanes 16-31 -> M = r+8, N = lane&15
  const int m0 = blockIdx.x * 16 + half * 8;
  #pragma unroll
  for (int r = 0; r < 8; ++r) {
    float v = c[r];
    Wh [(size_t)(m0 + r) * DOUT + n] = v;
    Whb[(size_t)(m0 + r) * DOUT + n] = f2bf(v);
  }
}

// ---------------------------------------------------------------------------
// Kernel 2: f1 = Wh . a[:128], f2 = Wh . a[128:]  (one wave per row)
// ---------------------------------------------------------------------------
__global__ __launch_bounds__(256) void k_proj_a(
    const float* __restrict__ Wh, const float* __restrict__ a,
    float* __restrict__ f1, float* __restrict__ f2)
{
  const int lane = threadIdx.x & 31, wave = threadIdx.x >> 5;
  const int row = blockIdx.x * 8 + wave;
  float s1 = 0.f, s2 = 0.f;
  #pragma unroll
  for (int c = lane; c < DOUT; c += 32) {
    float v = Wh[(size_t)row * DOUT + c];
    s1 += v * a[c];
    s2 += v * a[DOUT + c];
  }
  #pragma unroll
  for (int off = 16; off > 0; off >>= 1) {
    s1 += __shfl_xor(s1, off, 32);
    s2 += __shfl_xor(s2, off, 32);
  }
  if (lane == 0) { f1[row] = s1; f2[row] = s2; }
}

// ---------------------------------------------------------------------------
// Kernel 3: per-row masked softmax stats: m = max_j e_ij, sinv = 1/sum exp(e-m)
// (adj has self-loops so every row has >=1 valid entry). One wave per row.
// ---------------------------------------------------------------------------
__global__ __launch_bounds__(256) void k_stats(
    const float* __restrict__ adj, const float* __restrict__ f1,
    const float* __restrict__ f2, float* __restrict__ mrow,
    float* __restrict__ sinv)
{
  const int lane = threadIdx.x & 31, wave = threadIdx.x >> 5;
  const int row = blockIdx.x * 8 + wave;        // b*2048 + i
  const int b = row >> 11, i = row & (NN - 1);
  const float fi = f1[row];
  const float* __restrict__ ar = adj + (size_t)i * NN;
  const float* __restrict__ fb = f2 + (size_t)b * NN;
  float mx = -__builtin_inff();
  for (int j = lane; j < NN; j += 32)
    if (ar[j] > 0.f) {
      float e = fi + fb[j];
      e = e >= 0.f ? e : 0.2f * e;
      mx = fmaxf(mx, e);
    }
  #pragma unroll
  for (int off = 16; off > 0; off >>= 1) mx = fmaxf(mx, __shfl_xor(mx, off, 32));
  float s = 0.f;
  for (int j = lane; j < NN; j += 32)
    if (ar[j] > 0.f) {
      float e = fi + fb[j];
      e = e >= 0.f ? e : 0.2f * e;
      s += __expf(e - mx);
    }
  #pragma unroll
  for (int off = 16; off > 0; off >>= 1) s += __shfl_xor(s, off, 32);
  if (lane == 0) { mrow[row] = mx; sinv[row] = 1.f / s; }
}

// ---------------------------------------------------------------------------
// Kernel 4: fused attention GEMM, double-buffered.
// One block = (b, 16-row i-tile); 8 waves cover the 128 output columns.
// K = j streamed in 32-chunks; per iteration ONE barrier:
//   barrier -> issue next chunk (global loads, transposed LDS stage, coop P
//   with exp) into buf^1 -> compute frags + v_wmma_f32_16x16x32_bf16 on buf.
// Global latency + exp VALU of chunk t+1 overlap the WMMA of chunk t.
// ---------------------------------------------------------------------------
__global__ __launch_bounds__(256) void k_attn(
    const float* __restrict__ adj, const unsigned short* __restrict__ Whb,
    const float* __restrict__ f1, const float* __restrict__ f2,
    const float* __restrict__ mrow, const float* __restrict__ sinv,
    float* __restrict__ out)
{
  __shared__ unsigned short WhT[2][DOUT][40]; // [buf][n][k] transposed, padded
  __shared__ unsigned int   Pl[2][16][20];    // [buf][m][k/2] bf16 pairs, padded

  const int tid  = threadIdx.x;
  const int lane = tid & 31, wave = tid >> 5;
  const int half = lane >> 4, ml = lane & 15;
  const int b = blockIdx.x >> 7, itile = blockIdx.x & 127, i0 = itile * 16;
  const int n0 = wave * 16;

  // cooperative-P role: thread -> (row pm, k-pair pk)
  const int pm = tid >> 4, pk = tid & 15;
  const int prow = b * NN + i0 + pm;
  const float pf1 = f1[prow], pmv = mrow[prow], psv = sinv[prow];
  const float* __restrict__ ar = adj + (size_t)(i0 + pm) * NN;
  const float* __restrict__ fb = f2 + (size_t)b * NN;

  // staging role: thread -> (chunk row sr, 16-col slab sc0); coalesced 256B rows
  const int sr = tid >> 3, sc0 = (tid & 7) << 4;
  const unsigned short* __restrict__ wb = Whb + (size_t)(b * NN) * DOUT;

  auto stage = [&](int j0, int buf) {
    // 32x128 bf16 chunk of Wh, written transposed into LDS
    const unsigned short* src = wb + (size_t)(j0 + sr) * DOUT + sc0;
    union { v4u q[2]; unsigned short s[16]; } t;
    t.q[0] = *(const v4u*)src;
    t.q[1] = *(const v4u*)(src + 8);
    #pragma unroll
    for (int q = 0; q < 16; ++q) WhT[buf][sc0 + q][sr] = t.s[q];
    // cooperative P tile: 2 probabilities per thread, packed bf16 pair
    const int j = j0 + pk * 2;
    float e0 = pf1 + fb[j];     e0 = e0 >= 0.f ? e0 : 0.2f * e0;
    float e1 = pf1 + fb[j + 1]; e1 = e1 >= 0.f ? e1 : 0.2f * e1;
    float p0 = ar[j]     > 0.f ? __expf(e0 - pmv) * psv : 0.f;
    float p1 = ar[j + 1] > 0.f ? __expf(e1 - pmv) * psv : 0.f;
    Pl[buf][pm][pk] = (unsigned)f2bf(p0) | ((unsigned)f2bf(p1) << 16);
  };

  stage(0, 0);                      // prologue fill of buffer 0
  v8f c = {};
  for (int t = 0; t < NN / 32; ++t) {
    __syncthreads();                // publishes buf (t&1); protects buf (t&1)^1
    const int cur = t & 1;
    if (t + 1 < NN / 32) stage((t + 1) * 32, cur ^ 1);
    // A frag (16-bit A 16x32): lane M=ml; dwords [half*4..+3] = K 0-7/8-15,
    // dwords [half*4+8..+3] = K 16-23/24-31
    ABf A, Bm;
    A.q[0] = *(const v4u*)&Pl[cur][ml][half * 4];
    A.q[1] = *(const v4u*)&Pl[cur][ml][half * 4 + 8];
    // B frag (16-bit B 32x16): lane N=n0+ml; elems 0-7 K=half*8..+7, 8-15 K=16+half*8..+7
    const unsigned short* bp = &WhT[cur][n0 + ml][half * 8];
    Bm.q[0] = *(const v4u*)bp;
    Bm.q[1] = *(const v4u*)(bp + 16);
    c = __builtin_amdgcn_wmma_f32_16x16x32_bf16(false, A.v, false, Bm.v,
                                                (short)0, c, false, false);
  }
  const int om0 = i0 + half * 8, on = n0 + ml;
  float* __restrict__ ob = out + (size_t)b * NN * DOUT;
  #pragma unroll
  for (int r = 0; r < 8; ++r)
    ob[(size_t)(om0 + r) * DOUT + on] = c[r];
}

// ---------------------------------------------------------------------------
extern "C" void kernel_launch(void* const* d_in, const int* in_sizes, int n_in,
                              void* d_out, int out_size, void* d_ws, size_t ws_size,
                              hipStream_t stream) {
  (void)in_sizes; (void)n_in; (void)out_size; (void)ws_size;
  const float* h   = (const float*)d_in[0];   // (8,2048,256)
  const float* adj = (const float*)d_in[1];   // (2048,2048)
  const float* W   = (const float*)d_in[2];   // (256,128)
  const float* a   = (const float*)d_in[3];   // (256,)
  float* out = (float*)d_out;                 // (8,2048,128)

  float*          Wh  = (float*)d_ws;                                // 8.4 MB
  unsigned short* Whb = (unsigned short*)(Wh + (size_t)ROWS * DOUT); // 4.2 MB
  float*          f1  = (float*)(Whb + (size_t)ROWS * DOUT);
  float*          f2  = f1 + ROWS;
  float*          mr  = f2 + ROWS;
  float*          sv  = mr + ROWS;

  k_gemm_wh<<<ROWS / 16, 256, 0, stream>>>(h, W, Wh, Whb);
  k_proj_a <<<ROWS / 8, 256, 0, stream>>>(Wh, a, f1, f2);
  k_stats  <<<ROWS / 8, 256, 0, stream>>>(adj, f1, f2, mr, sv);
  k_attn   <<<NB * (NN / 16), 256, 0, stream>>>(adj, Whb, f1, f2, mr, sv, out);
}
